// FA_49177375539263
// MI455X (gfx1250) — compile-verified, matrix operand order checked by reading, not run on hardware
//
#include <hip/hip_runtime.h>

#define DIMC 112
#define DHD  28
#define NHEAD 4
#define NTOK 64
#define BSPD 8
#define IMG  128

typedef __attribute__((ext_vector_type(16))) __bf16       v16bf;
typedef __attribute__((ext_vector_type(8)))  float        v8f;
typedef __attribute__((ext_vector_type(4)))  float        v4f;
typedef __attribute__((ext_vector_type(4)))  unsigned int v4u;

union FragL { v4u q[2]; v16bf v; };           // LDS fragment: 2 x b128
union FragB { __bf16 b[16]; v16bf v; };       // converted global fragment
union FB    { float f; unsigned int u; };

static __device__ __forceinline__ unsigned short f2bfu(float x) {
  union { __bf16 b; unsigned short u; } t;
  t.b = (__bf16)x;                            // hardware cvt f32 -> bf16
  return t.u;
}
static __device__ __forceinline__ float bfu2f(unsigned short h) {
  FB t; t.u = ((unsigned int)h) << 16; return t.f;
}

// Fragment from an LDS array laid out [row][k] (k contiguous, bf16 bits).
// CDNA5 16-bit 16x32 operand layout => per lane two contiguous 8-elem runs:
//   k = k0+kh .. +7   and   k = k0+16+kh .. +7,  kh = (lane>=16) ? 8 : 0
// Each run is 16 bytes -> one ds_read_b128.
static __device__ __forceinline__ v16bf load_frag_lds(const unsigned short* S,
                                                      int r0, int ld, int k0,
                                                      int lane) {
  int r  = r0 + (lane & 15);
  int kh = (lane >> 4) << 3;
  const unsigned short* base = S + r * ld + k0 + kh;
  FragL f;
  f.q[0] = *(const v4u*)(base);
  f.q[1] = *(const v4u*)(base + 16);
  return f.v;
}

// B fragment gathered from a global fp32 weight matrix W[n][k] (k contiguous),
// converted to bf16 with hardware cvt. Runs are 8-aligned and ld % 8 == 0, so
// each 8-float run is either fully valid or fully out of bounds (-> zeros).
static __device__ __forceinline__ v16bf load_bfrag_g(const float* __restrict__ W,
                                                     int n0, int ld, int k0,
                                                     int lane, int nmax) {
  int n = n0 + (lane & 15); if (n > nmax) n = nmax;
  int kh = (lane >> 4) << 3;
  const float* row = W + n * ld;
  int ka = k0 + kh;
  int kb = k0 + 16 + kh;
  v4f a0 = {}, a1 = {}, b0 = {}, b1 = {};
  if (ka + 8 <= ld) { a0 = *(const v4f*)(row + ka); a1 = *(const v4f*)(row + ka + 4); }
  if (kb + 8 <= ld) { b0 = *(const v4f*)(row + kb); b1 = *(const v4f*)(row + kb + 4); }
  FragB f;
#pragma unroll
  for (int e = 0; e < 4; ++e) {
    f.b[e]      = (__bf16)a0[e];
    f.b[e + 4]  = (__bf16)a1[e];
    f.b[e + 8]  = (__bf16)b0[e];
    f.b[e + 12] = (__bf16)b1[e];
  }
  return f.v;
}

static __device__ __forceinline__ v8f wmma_bf16(v16bf a, v16bf b, v8f c) {
  return __builtin_amdgcn_wmma_f32_16x16x32_bf16(false, a, false, b,
                                                 (short)0, c, false, false);
}

__global__ __launch_bounds__(256) void FA_win_attn_kernel(
    const float* __restrict__ x,     const float* __restrict__ w_qk,
    const float* __restrict__ w_v,   const float* __restrict__ w_out,
    const float* __restrict__ b_out, const float* __restrict__ w_pq,
    const float* __restrict__ b_pq,  const float* __restrict__ w_pk,
    const float* __restrict__ b_pk,  const float* __restrict__ w_m1,
    const float* __restrict__ w_m2a, const float* __restrict__ w_m2b,
    float* __restrict__ out) {
  __shared__ __attribute__((aligned(16))) unsigned short Xbf[NTOK * 128];
  __shared__ __attribute__((aligned(16))) unsigned short AO [NTOK * 128];
  __shared__ __attribute__((aligned(16))) unsigned short Qh [NTOK * 32];
  __shared__ __attribute__((aligned(16))) unsigned short Kh [NTOK * 32];
  __shared__ __attribute__((aligned(16))) unsigned short Vt [32 * NTOK];
  __shared__ __attribute__((aligned(16))) float simf[NTOK * 64];  // also bf16 attn view
  __shared__ float sqA[64], skA[64], th1[64], th2[64];
  __shared__ float thsc;

  const int tid  = threadIdx.x;
  const int lane = tid & 31;
  const int wave = tid >> 5;
  const int blk  = blockIdx.x;
  const int Bidx = blk & 7;      // spectral index
  const int win  = blk >> 3;     // window index (wh*16 + ww)
  const int wh   = win >> 4;
  const int ww   = win & 15;

  if (tid == 0) {
    __builtin_prefetch(w_qk, 0, 0);
    __builtin_prefetch(w_v, 0, 0);
    __builtin_prefetch(w_out, 0, 0);
    __builtin_prefetch(w_m2a, 0, 0);
  }

  // ---- gather rolled window into LDS (bf16); token-fastest for coalescing ----
  for (int idx = tid; idx < NTOK * DIMC; idx += 256) {
    int c = idx >> 6;            // 0..111
    int t = idx & 63;
    int b0 = t >> 3, b1 = t & 7;
    int hh = (wh * 8 + b0 + 124) & 127;   // roll(+4): src = pos-4 mod 128
    int wc = (ww * 8 + b1 + 124) & 127;
    float v = x[(((size_t)c * BSPD + Bidx) * IMG + hh) * IMG + wc];
    Xbf[t * 128 + c] = f2bfu(v);
  }
  for (int idx = tid; idx < NTOK * 16; idx += 256)
    Xbf[(idx >> 4) * 128 + 112 + (idx & 15)] = 0;
  for (int idx = tid; idx < NTOK * 128; idx += 256) AO[idx] = 0;
  for (int idx = tid; idx < NTOK * 32; idx += 256) {
    Qh[idx] = 0; Kh[idx] = 0; Vt[idx] = 0;
  }
  __syncthreads();

  for (int h = 0; h < NHEAD; ++h) {
    // ---- per-head QKV projection: 4 M-tiles x 6 N-tiles, K = 4x32 ----
    for (int tt = wave; tt < 24; tt += 8) {
      int mt = tt & 3;
      int nidx = tt >> 2;          // 0..5
      int kind = nidx >> 1;        // 0=q 1=k 2=v
      int sub  = nidx & 1;
      v8f acc = {};
#pragma unroll
      for (int ks = 0; ks < 4; ++ks) {
        v16bf a = load_frag_lds(Xbf, mt * 16, 128, ks * 32, lane);
        const float* W = (kind == 2) ? w_v : w_qk;
        int nbase = h * DHD + sub * 16 + ((kind == 1) ? DIMC : 0);
        int nmax  = (kind == 2) ? (DIMC - 1) : (2 * DIMC - 1);
        v16bf b = load_bfrag_g(W, nbase, DIMC, ks * 32, lane, nmax);
        acc = wmma_bf16(a, b, acc);
      }
      int n = sub * 16 + (lane & 15);
      if (n < DHD) {
        int mb = mt * 16 + ((lane >> 4) << 3);
#pragma unroll
        for (int r = 0; r < 8; ++r) {
          unsigned short v = f2bfu(acc[r]);
          int m = mb + r;
          if (kind == 0)      Qh[m * 32 + n] = v;
          else if (kind == 1) Kh[m * 32 + n] = v;
          else                Vt[n * 64 + m] = v;   // transposed store
        }
      }
    }
    __syncthreads();

    // ---- sim = q @ k^T : 16 tiles, single K step (d padded 28->32) ----
    for (int tt = wave; tt < 16; tt += 8) {
      int mt = tt & 3, nt = tt >> 2;
      v16bf a = load_frag_lds(Qh, mt * 16, 32, 0, lane);
      v16bf b = load_frag_lds(Kh, nt * 16, 32, 0, lane);
      v8f acc = {};
      acc = wmma_bf16(a, b, acc);
      int j  = nt * 16 + (lane & 15);
      int mb = mt * 16 + ((lane >> 4) << 3);
#pragma unroll
      for (int r = 0; r < 8; ++r) simf[(mb + r) * 64 + j] = acc[r];
    }
    // sigma projections (overlap with sim GEMM; reads only Qh/Kh)
    if (tid < 64) {
      float aq = b_pq[0], ak = b_pk[0];
      for (int d = 0; d < DHD; ++d) {
        aq += bfu2f(Qh[tid * 32 + d]) * w_pq[d];
        ak += bfu2f(Kh[tid * 32 + d]) * w_pk[d];
      }
      sqA[tid] = aq; skA[tid] = ak;
    }
    __syncthreads();

    // ---- threshold MLP: theta1 = sim_norm @ w_m1 (diag zeroed) ----
    if (tid < 64) {
      float t1 = 0.f;
      for (int j = 0; j < 64; ++j) t1 += simf[tid * 64 + j] * w_m1[j];
      t1 -= simf[tid * 64 + tid] * w_m1[tid];
      th1[tid] = t1;
    }
    __syncthreads();
    if (tid < 64) {
      float t2 = 0.f;
      for (int n = 0; n < 64; ++n) t2 += th1[n] * w_m2a[tid * 64 + n];
      th2[tid] = (t2 > 0.f) ? t2 : 0.1f * t2;   // leaky relu
    }
    __syncthreads();
    if (tid == 0) {
      float t = 0.f;
      for (int m = 0; m < 64; ++m) t += th2[m] * w_m2b[m];
      thsc = t;
    }
    __syncthreads();

    // ---- Sigma scale, softmax, threshold mask; write bf16 attn in-place ----
    if (tid < 64) {
      float row[64];
      float sq_ = sqA[tid], th = thsc;
      float mx = -3.4e38f;
      for (int j = 0; j < 64; ++j) {
        float s = simf[tid * 64 + j] * sq_ * skA[j];
        row[j] = s;
        mx = fmaxf(mx, s);
      }
      unsigned int mlo = 0, mhi = 0;
      float sum = 0.f;
      for (int j = 0; j < 64; ++j) {
        float s = row[j];
        if (s > th) { if (j < 32) mlo |= 1u << j; else mhi |= 1u << (j - 32); }
        float ev = __expf(s - mx);
        row[j] = ev; sum += ev;
      }
      float inv = 1.f / sum;
      unsigned int* attnw = (unsigned int*)simf;   // packed bf16x2, row = 64 dwords
      for (int j = 0; j < 64; j += 2) {
        int b0_ = (j < 32) ? ((mlo >> j) & 1) : ((mhi >> (j - 32)) & 1);
        int b1_ = ((j + 1) < 32) ? ((mlo >> (j + 1)) & 1) : ((mhi >> (j + 1 - 32)) & 1);
        unsigned int lo = f2bfu(b0_ ? row[j] * inv : 0.f);
        unsigned int hi = f2bfu(b1_ ? row[j + 1] * inv : 0.f);
        attnw[tid * 64 + (j >> 1)] = lo | (hi << 16);
      }
    }
    __syncthreads();

    // ---- out_h = attn @ v : 4 M-tiles x 2 N-tiles, K = 2x32 ----
    {
      const unsigned short* attnr = (const unsigned short*)simf;  // row stride 128
      for (int tt = wave; tt < 8; tt += 8) {
        int mt = tt & 3, nt = tt >> 2;
        v8f acc = {};
#pragma unroll
        for (int ks = 0; ks < 2; ++ks) {
          v16bf a = load_frag_lds(attnr, mt * 16, 128, ks * 32, lane);
          v16bf b = load_frag_lds(Vt,    nt * 16,  64, ks * 32, lane);
          acc = wmma_bf16(a, b, acc);
        }
        int d = nt * 16 + (lane & 15);
        if (d < DHD) {
          int mb = mt * 16 + ((lane >> 4) << 3);
#pragma unroll
          for (int r = 0; r < 8; ++r)
            AO[(mb + r) * 128 + h * DHD + d] = f2bfu(acc[r]);
        }
      }
    }
    __syncthreads();
  }

  // ---- output projection: 4 M-tiles x 7 N-tiles, K = 4x32 ----
  for (int tt = wave; tt < 28; tt += 8) {
    int mt = tt & 3, nt = tt >> 2;
    v8f acc = {};
#pragma unroll
    for (int ks = 0; ks < 4; ++ks) {
      v16bf a = load_frag_lds(AO, mt * 16, 128, ks * 32, lane);
      v16bf b = load_bfrag_g(w_out, nt * 16, DIMC, ks * 32, lane, DIMC - 1);
      acc = wmma_bf16(a, b, acc);
    }
    int o = nt * 16 + (lane & 15);
    float bias = b_out[o];
    int mb = mt * 16 + ((lane >> 4) << 3);
#pragma unroll
    for (int r = 0; r < 8; ++r) {
      int m = mb + r;
      int b0 = m >> 3, b1 = m & 7;
      int hh = (wh * 8 + b0 + 124) & 127;   // roll(-4) == -4 mod 128
      int wc = (ww * 8 + b1 + 124) & 127;
      out[(((size_t)o * BSPD + Bidx) * IMG + hh) * IMG + wc] = acc[r] + bias;
    }
  }
}

extern "C" void kernel_launch(void* const* d_in, const int* in_sizes, int n_in,
                              void* d_out, int out_size, void* d_ws, size_t ws_size,
                              hipStream_t stream) {
  (void)in_sizes; (void)n_in; (void)out_size; (void)d_ws; (void)ws_size;
  const float* x     = (const float*)d_in[0];
  const float* w_qk  = (const float*)d_in[1];
  const float* w_v   = (const float*)d_in[2];
  const float* w_out = (const float*)d_in[3];
  const float* b_out = (const float*)d_in[4];
  const float* w_pq  = (const float*)d_in[5];
  const float* b_pq  = (const float*)d_in[6];
  const float* w_pk  = (const float*)d_in[7];
  const float* b_pk  = (const float*)d_in[8];
  const float* w_m1  = (const float*)d_in[9];
  const float* w_m2a = (const float*)d_in[10];
  const float* w_m2b = (const float*)d_in[11];
  float* out = (float*)d_out;

  // one block per (window, spectral-B): 256 windows * 8 = 2048 blocks
  FA_win_attn_kernel<<<dim3(2048), dim3(256), 0, stream>>>(
      x, w_qk, w_v, w_out, b_out, w_pq, b_pq, w_pk, b_pk,
      w_m1, w_m2a, w_m2b, out);
}
